// CombinedHiddenGCVAE_38886633898290
// MI455X (gfx1250) — compile-verified, matrix operand order checked by reading, not run on hardware
//
#include <hip/hip_runtime.h>
#include <hip/hip_bf16.h>

#define Nn   50000
#define Ee   800000
#define FEAT 64
#define COND 32
#define HID  64
#define LAT  32

typedef __attribute__((ext_vector_type(16))) __bf16 v16bf;
typedef __attribute__((ext_vector_type(8)))  float  v8f;

union Frag16 {
    v16bf        v;
    unsigned int u[8];
    uint4        q[2];
};

// ---------- helpers ----------

__device__ __forceinline__ unsigned int f2bf(float f) {
    // round-to-nearest-even f32 -> bf16 (upper 16 bits)
    unsigned int u = __float_as_uint(f);
    unsigned int r = u + 0x7FFFu + ((u >> 16) & 1u);
    return r >> 16;
}

__device__ __forceinline__ unsigned int pcg(unsigned int v) {
    unsigned int s = v * 747796405u + 2891336453u;
    unsigned int w = ((s >> ((s >> 28) + 4u)) ^ s) * 277803737u;
    return (w >> 22) ^ w;
}

__device__ __forceinline__ float gauss_hash(unsigned int idx) {
    unsigned int h1 = pcg(idx * 2u + 1u);
    unsigned int h2 = pcg(idx * 2u + 2u);
    float u1 = (float)(h1 >> 8) * (1.0f / 16777216.0f) + (0.5f / 16777216.0f);
    float u2 = (float)(h2 >> 8) * (1.0f / 16777216.0f);
    float r  = sqrtf(-2.0f * __logf(u1));
    return r * __cosf(6.28318530718f * u2);
}

// ---------- edge preprocessing ----------

__global__ void k_edge_cvt(const long long* __restrict__ e64,
                           int* __restrict__ s32, int* __restrict__ d32) {
    int e = blockIdx.x * blockDim.x + threadIdx.x;
    if (e < Ee) {
        s32[e] = (int)e64[e];
        d32[e] = (int)e64[Ee + e];
    }
}

__global__ void k_deg_init(float* __restrict__ deg) {
    int i = blockIdx.x * blockDim.x + threadIdx.x;
    if (i < Nn) deg[i] = 1.0f;           // self loop
}

__global__ void k_deg_count(const int* __restrict__ dst, float* __restrict__ deg) {
    int e = blockIdx.x * blockDim.x + threadIdx.x;
    if (e < Ee) atomicAdd(&deg[dst[e]], 1.0f);
}

__global__ void k_dinv(const float* __restrict__ deg, float* __restrict__ dinv) {
    int i = blockIdx.x * blockDim.x + threadIdx.x;
    if (i < Nn) dinv[i] = rsqrtf(deg[i]);
}

__global__ void k_coef(const int* __restrict__ s32, const int* __restrict__ d32,
                       const float* __restrict__ dinv, float* __restrict__ coef) {
    int e = blockIdx.x * blockDim.x + threadIdx.x;
    if (e < Ee) coef[e] = dinv[s32[e]] * dinv[d32[e]];
}

// ---------- concat + bf16 cast (encoder input: [feature | condition]) ----------

__global__ void k_concat_enc(const float* __restrict__ feat, const float* __restrict__ cond,
                             unsigned short* __restrict__ xc) {
    int idx = blockIdx.x * blockDim.x + threadIdx.x;     // N * 96
    if (idx >= Nn * (FEAT + COND)) return;
    int i = idx / (FEAT + COND);
    int c = idx - i * (FEAT + COND);
    float v = (c < FEAT) ? feat[i * FEAT + c] : cond[i * COND + (c - FEAT)];
    xc[idx] = (unsigned short)f2bf(v);
}

// ---------- pack f32 weights into pre-swizzled bf16 WMMA B fragments ----------
// layout: [(tileN * ksteps + ks)][lane][v]  (dwords), so a wave's k-step B load
// is two coalesced b128 loads per lane.

__global__ void k_pack_w(const float* __restrict__ W, unsigned int* __restrict__ BW,
                         int K, int Nout) {
    int idx    = blockIdx.x * blockDim.x + threadIdx.x;
    int ksteps = K >> 5;
    int total  = (Nout >> 4) * ksteps * 32 * 8;
    if (idx >= total) return;
    int v    = idx & 7;
    int lane = (idx >> 3) & 31;
    int t    = idx >> 8;                // tile * ksteps + ks
    int tile = t / ksteps;
    int ks   = t - tile * ksteps;
    // ISA 16-bit fragment layout: K = 16*(v>>2) + 8*(lane>>4) + 2*(v&3)
    int k = (ks << 5) + ((v >> 2) << 4) + ((lane >> 4) << 3) + ((v & 3) << 1);
    int n = (tile << 4) + (lane & 15);
    unsigned int lo = f2bf(W[(size_t)k * Nout + n]);
    unsigned int hi = f2bf(W[(size_t)(k + 1) * Nout + n]);
    BW[idx] = lo | (hi << 16);
}

// ---------- WMMA GEMM: C[M,Nout] = A_bf16[M,K] @ B(pre-swizzled) ----------
// one wave32 per 16x16 tile; KSTEPS fully unrolled -> back-to-back wmma

template <int KSTEPS>
__global__ void k_gemm_wmma(const unsigned short* __restrict__ A,
                            const unsigned int* __restrict__ BW,
                            float* __restrict__ C,
                            int M, int Nout) {
    const int K = KSTEPS * 32;
    int wave   = (blockIdx.x * blockDim.x + threadIdx.x) >> 5;
    int lane   = threadIdx.x & 31;
    int tilesN = Nout >> 4;
    int tiles  = (M >> 4) * tilesN;
    if (wave >= tiles) return;                 // whole-wave uniform -> EXEC all 1s below
    int tmt = wave / tilesN;
    int tnt = wave - tmt * tilesN;
    int tm  = tmt << 4;
    int tn  = tnt << 4;

    int half8 = (lane >> 4) << 3;              // +8 K offset for upper half-wave
    int l15   = lane & 15;
    const unsigned short* arow = A + (size_t)(tm + l15) * K;

    v8f acc = {};
#pragma unroll
    for (int ks = 0; ks < KSTEPS; ++ks) {
        Frag16 a, b;
        // A fragment: two contiguous 16B runs (K = half8..+7 and K+16..+7)
        a.q[0] = *(const uint4*)(arow + (ks << 5) + half8);
        a.q[1] = *(const uint4*)(arow + (ks << 5) + 16 + half8);
        // B fragment: pre-swizzled, lane-major contiguous 32B
        const uint4* bp = (const uint4*)(BW + ((size_t)(tnt * KSTEPS + ks) * 32 + lane) * 8);
        b.q[0] = bp[0];
        b.q[1] = bp[1];
        acc = __builtin_amdgcn_wmma_f32_16x16x32_bf16(
            /*neg_a=*/false, a.v, /*neg_b=*/false, b.v,
            /*c_mod=*/(short)0, acc, /*reuse_a=*/false, /*reuse_b=*/false);
    }
#pragma unroll
    for (int r = 0; r < 8; ++r) {
        int row = tm + r + half8;              // C layout: VGPR r -> M = r + 8*half
        C[(size_t)row * Nout + tn + l15] = acc[r];
    }
}

// ---------- aggregation: self-loop + bias init, then edge scatter ----------

__global__ void k_agg_init(const float* __restrict__ xw, const float* __restrict__ dinv,
                           const float* __restrict__ bias, float* __restrict__ agg,
                           int fshift /* log2(F) */) {
    int idx = blockIdx.x * blockDim.x + threadIdx.x;     // N * F
    int F   = 1 << fshift;
    if (idx >= Nn * F) return;
    int i = idx >> fshift;
    int f = idx & (F - 1);
    float di = dinv[i];
    agg[idx] = xw[idx] * (di * di) + bias[f];            // 1/deg == dinv^2
}

__global__ void k_agg_edges(const int* __restrict__ s32, const int* __restrict__ d32,
                            const float* __restrict__ coef,
                            const float* __restrict__ xw, float* __restrict__ agg,
                            int F, int chunkShift /* log2(F/4) */) {
    int idx = blockIdx.x * blockDim.x + threadIdx.x;     // E * F/4
    int e   = idx >> chunkShift;
    if (e >= Ee) return;
    int c   = idx & ((1 << chunkShift) - 1);
    int s   = s32[e];
    int d   = d32[e];
    float cf = coef[e];
    __builtin_prefetch(agg + (size_t)d * F + 4 * c, 1, 0);   // global_prefetch_b8
    const float4 m = *(const float4*)(xw + (size_t)s * F + 4 * c);
    float* p = agg + (size_t)d * F + 4 * c;
    atomicAdd(p + 0, m.x * cf);
    atomicAdd(p + 1, m.y * cf);
    atomicAdd(p + 2, m.z * cf);
    atomicAdd(p + 3, m.w * cf);
}

// ---------- relu + bf16 cast ----------

__global__ void k_relu_cast(const float* __restrict__ h, unsigned short* __restrict__ hb, int n) {
    int i = blockIdx.x * blockDim.x + threadIdx.x;
    if (i < n) hb[i] = (unsigned short)f2bf(fmaxf(h[i], 0.0f));
}

// ---------- reparameterization + decoder input concat ----------

__global__ void k_reparam(const float* __restrict__ mean, const float* __restrict__ logvar,
                          const float* __restrict__ cond, float* __restrict__ z_out,
                          unsigned short* __restrict__ xcd) {
    int idx = blockIdx.x * blockDim.x + threadIdx.x;     // N * LAT
    if (idx >= Nn * LAT) return;
    int i = idx / LAT;
    int l = idx - i * LAT;
    float noise = gauss_hash((unsigned int)idx);
    float z = noise * expf(0.5f * logvar[idx]) + mean[idx];
    z_out[idx] = z;
    xcd[(size_t)i * (LAT + COND) + l]       = (unsigned short)f2bf(z);
    xcd[(size_t)i * (LAT + COND) + LAT + l] = (unsigned short)f2bf(cond[i * COND + l]);
}

// ---------- launch ----------

static inline dim3 g1(long long n, int b) { return dim3((unsigned)((n + b - 1) / b)); }

extern "C" void kernel_launch(void* const* d_in, const int* in_sizes, int n_in,
                              void* d_out, int out_size, void* d_ws, size_t ws_size,
                              hipStream_t stream) {
    const float*     feature   = (const float*)d_in[0];
    const float*     condition = (const float*)d_in[1];
    const long long* edge      = (const long long*)d_in[2];
    const float*     enc_i2h_W = (const float*)d_in[3];
    const float*     enc_i2h_b = (const float*)d_in[4];
    const float*     enc_mn_W  = (const float*)d_in[5];
    const float*     enc_mn_b  = (const float*)d_in[6];
    const float*     enc_lv_W  = (const float*)d_in[7];
    const float*     enc_lv_b  = (const float*)d_in[8];
    const float*     dec_i2h_W = (const float*)d_in[9];
    const float*     dec_i2h_b = (const float*)d_in[10];
    const float*     dec_out_W = (const float*)d_in[11];
    const float*     dec_out_b = (const float*)d_in[12];

    // output slices: (z, mean, logvar, out) flat
    float* z_out  = (float*)d_out;
    float* mn_out = z_out  + (size_t)Nn * LAT;
    float* lv_out = mn_out + (size_t)Nn * LAT;
    float* oo_out = lv_out + (size_t)Nn * LAT;

    // workspace carve (256B aligned)
    size_t off = 0;
    auto carve = [&](size_t bytes) -> void* {
        void* p = (char*)d_ws + off;
        off += (bytes + 255) & ~(size_t)255;
        return p;
    };
    float*          deg   = (float*)carve((size_t)Nn * 4);
    float*          dinv  = (float*)carve((size_t)Nn * 4);
    int*            s32   = (int*)carve((size_t)Ee * 4);
    int*            d32   = (int*)carve((size_t)Ee * 4);
    float*          coef  = (float*)carve((size_t)Ee * 4);
    unsigned short* xce   = (unsigned short*)carve((size_t)Nn * (FEAT + COND) * 2);
    unsigned short* xcd   = (unsigned short*)carve((size_t)Nn * (LAT + COND) * 2);
    float*          xw    = (float*)carve((size_t)Nn * HID * 4);
    float*          hf    = (float*)carve((size_t)Nn * HID * 4);
    unsigned short* hbf   = (unsigned short*)carve((size_t)Nn * HID * 2);
    unsigned int*   bw    = (unsigned int*)carve((size_t)16384);   // packed weights (<=12KB)
    (void)ws_size; (void)n_in; (void)in_sizes; (void)out_size;

    const int B = 256;
    auto gemm = [&](const unsigned short* A, const float* W, float* C, int K, int Nout) {
        int ksteps = K >> 5;
        int pack_n = (Nout >> 4) * ksteps * 256;
        k_pack_w<<<g1(pack_n, B), B, 0, stream>>>(W, bw, K, Nout);
        int tiles = (Nn >> 4) * (Nout >> 4);
        if (ksteps == 3)
            k_gemm_wmma<3><<<g1((long long)tiles * 32, B), B, 0, stream>>>(A, bw, C, Nn, Nout);
        else
            k_gemm_wmma<2><<<g1((long long)tiles * 32, B), B, 0, stream>>>(A, bw, C, Nn, Nout);
    };
    auto agg = [&](const float* xwp, const float* bias, float* out, int F) {
        int fshift = (F == 64) ? 6 : 5;
        int cshift = (F == 64) ? 4 : 3;
        k_agg_init<<<g1((long long)Nn * F, B), B, 0, stream>>>(xwp, dinv, bias, out, fshift);
        k_agg_edges<<<g1((long long)Ee * (F >> 2), B), B, 0, stream>>>(s32, d32, coef, xwp, out, F, cshift);
    };

    // edge preprocessing + degree normalization
    k_edge_cvt <<<g1(Ee, B), B, 0, stream>>>(edge, s32, d32);
    k_deg_init <<<g1(Nn, B), B, 0, stream>>>(deg);
    k_deg_count<<<g1(Ee, B), B, 0, stream>>>(d32, deg);
    k_dinv     <<<g1(Nn, B), B, 0, stream>>>(deg, dinv);
    k_coef     <<<g1(Ee, B), B, 0, stream>>>(s32, d32, dinv, coef);

    // encoder hidden: h = relu(gcn(concat(feature, condition), enc_i2h))
    k_concat_enc<<<g1((long long)Nn * (FEAT + COND), B), B, 0, stream>>>(feature, condition, xce);
    gemm(xce, enc_i2h_W, xw, FEAT + COND, HID);
    agg(xw, enc_i2h_b, hf, HID);
    k_relu_cast<<<g1((long long)Nn * HID, B), B, 0, stream>>>(hf, hbf, Nn * HID);

    // mean / logvar heads (aggregate straight into d_out slices)
    gemm(hbf, enc_mn_W, xw, HID, LAT);
    agg(xw, enc_mn_b, mn_out, LAT);
    gemm(hbf, enc_lv_W, xw, HID, LAT);
    agg(xw, enc_lv_b, lv_out, LAT);

    // z = noise * exp(0.5*logvar) + mean ; build decoder input [z | condition] in bf16
    k_reparam<<<g1((long long)Nn * LAT, B), B, 0, stream>>>(mn_out, lv_out, condition, z_out, xcd);

    // decoder hidden: hd = relu(gcn([z|cond], dec_i2h))
    gemm(xcd, dec_i2h_W, xw, LAT + COND, HID);
    agg(xw, dec_i2h_b, hf, HID);
    k_relu_cast<<<g1((long long)Nn * HID, B), B, 0, stream>>>(hf, hbf, Nn * HID);

    // output: out = gcn(hd, dec_out)
    gemm(hbf, dec_out_W, xw, HID, FEAT);
    agg(xw, dec_out_b, oo_out, FEAT);
}
